// FmriEncoder_MoETransformer_77360950935852
// MI455X (gfx1250) — matrix operand order repaired and checked by reading, not compiled, not run on hardware
//
#include <hip/hip_runtime.h>
#include <cstdint>
#include <cstddef>

typedef __bf16 bf16;
typedef __attribute__((ext_vector_type(16))) __bf16 v16bf;
typedef __attribute__((ext_vector_type(8)))  float  v8f;
typedef __attribute__((ext_vector_type(4)))  __bf16 v4bf;

#define BT_TOK 16384
#define DIM    1024
#define NEXP   8
#define FDIM   4096
#define TOPK   2
#define NPAIR  (BT_TOK * TOPK)

union FragU { uint4 q[2]; v16bf v; };

// ---- CDNA5 async global->LDS staging (ISA 10.x / 15.18, tracked by ASYNCcnt) ----
__device__ __forceinline__ void async_ld_b128(unsigned lds_off, const void* gptr) {
    asm volatile("global_load_async_to_lds_b128 %0, %1, off"
                 :: "v"(lds_off), "v"((unsigned long long)(uintptr_t)gptr)
                 : "memory");
}
__device__ __forceinline__ void wait_async0() {
    asm volatile("s_wait_asynccnt 0" ::: "memory");
}
__device__ __forceinline__ unsigned lds_off_of(const void* p) {
    // LDS aperture occupies the upper address bits; low 32 bits are the LDS byte offset.
    return (unsigned)(uintptr_t)p;
}

// ---------------- utility kernels ----------------

__global__ void init_counts_kernel(int* counts, int* cursor) {
    int i = threadIdx.x;
    if (i < NEXP) { counts[i] = 0; cursor[i] = 0; }
}

__global__ void cvt_f32_to_bf16(const float* __restrict__ in, bf16* __restrict__ out, int n4) {
    int i = blockIdx.x * blockDim.x + threadIdx.x;
    if (i < n4) {
        float4 f = ((const float4*)in)[i];
        v4bf o;
        o.x = (bf16)f.x; o.y = (bf16)f.y; o.z = (bf16)f.z; o.w = (bf16)f.w;
        ((v4bf*)out)[i] = o;
    }
}

// fp32 [E][K][N] row-major  ->  bf16 [E][N][K]  (per-expert transpose, LDS-tiled 32x32)
__global__ void transpose_cvt_kernel(const float* __restrict__ in, bf16* __restrict__ out,
                                     int K, int N) {
    const int e  = blockIdx.z;
    const int n0 = blockIdx.x * 32;
    const int k0 = blockIdx.y * 32;
    const float* in_e = in + (size_t)e * (size_t)K * (size_t)N;
    bf16* out_e = out + (size_t)e * (size_t)K * (size_t)N;

    __shared__ float t[32][33];
    const int tid = threadIdx.x;
    const int r  = tid >> 3;         // 0..31
    const int c4 = (tid & 7) * 4;    // 0..28

    float4 f = *(const float4*)&in_e[(size_t)(k0 + r) * N + n0 + c4];
    t[r][c4 + 0] = f.x; t[r][c4 + 1] = f.y; t[r][c4 + 2] = f.z; t[r][c4 + 3] = f.w;
    __syncthreads();

    v4bf o;
    o.x = (bf16)t[c4 + 0][r];
    o.y = (bf16)t[c4 + 1][r];
    o.z = (bf16)t[c4 + 2][r];
    o.w = (bf16)t[c4 + 3][r];
    *(v4bf*)&out_e[(size_t)(n0 + r) * K + k0 + c4] = o;
}

// ---------------- gating: logits -> softmax -> top2 (one wave32 per token) ----------------
__global__ void gate_topk_kernel(const float* __restrict__ x, const float* __restrict__ gw,
                                 int* __restrict__ tok_e, float* __restrict__ tok_w,
                                 int* __restrict__ counts) {
    int wave = threadIdx.x >> 5, lane = threadIdx.x & 31;
    int t = blockIdx.x * 8 + wave;
    if (t >= BT_TOK) return;

    float acc[NEXP];
#pragma unroll
    for (int e = 0; e < NEXP; ++e) acc[e] = 0.0f;
    const float* xr = x + (size_t)t * DIM;
    for (int d = lane; d < DIM; d += 32) {
        float xv = xr[d];
#pragma unroll
        for (int e = 0; e < NEXP; ++e) acc[e] += xv * gw[e * DIM + d];
    }
#pragma unroll
    for (int e = 0; e < NEXP; ++e)
        for (int off = 16; off > 0; off >>= 1) acc[e] += __shfl_xor(acc[e], off, 32);

    if (lane == 0) {
        float m = acc[0];
#pragma unroll
        for (int e = 1; e < NEXP; ++e) m = fmaxf(m, acc[e]);
        float p[NEXP], s = 0.0f;
#pragma unroll
        for (int e = 0; e < NEXP; ++e) { p[e] = expf(acc[e] - m); s += p[e]; }
        float inv = 1.0f / s;
#pragma unroll
        for (int e = 0; e < NEXP; ++e) p[e] *= inv;
        int e1 = 0;
#pragma unroll
        for (int e = 1; e < NEXP; ++e) if (p[e] > p[e1]) e1 = e;
        int e2 = (e1 == 0) ? 1 : 0;
#pragma unroll
        for (int e = 0; e < NEXP; ++e) if (e != e1 && p[e] > p[e2]) e2 = e;
        tok_e[2 * t]     = e1; tok_w[2 * t]     = p[e1];
        tok_e[2 * t + 1] = e2; tok_w[2 * t + 1] = p[e2];
        atomicAdd(&counts[e1], 1);
        atomicAdd(&counts[e2], 1);
    }
}

__global__ void prefix_kernel(const int* __restrict__ counts, int* __restrict__ offsets) {
    if (threadIdx.x == 0) {
        int s = 0;
        for (int e = 0; e < NEXP; ++e) { offsets[e] = s; s += counts[e]; }
        offsets[NEXP] = s;
    }
}

__global__ void build_pairs_kernel(const int* __restrict__ tok_e, const float* __restrict__ tok_w,
                                   const int* __restrict__ offsets, int* __restrict__ cursor,
                                   int* __restrict__ pair_token, int* __restrict__ pair_slot) {
    int t = blockIdx.x * blockDim.x + threadIdx.x;
    if (t >= BT_TOK) return;
#pragma unroll
    for (int k = 0; k < TOPK; ++k) {
        int e = tok_e[2 * t + k];
        int pos = atomicAdd(&cursor[e], 1);
        int row = offsets[e] + pos;
        pair_token[row] = t;
        pair_slot[2 * t + k] = row;
    }
}

// ---------------- WMMA expert GEMM ----------------
// A: [rows][K] bf16 row-major (MODE1: gather token rows of xb; MODE2: compacted h rows)
// Bt: [E][N][K] bf16 (transposed weights) -> B fragments contiguous along K
// MODE1 out: h bf16 = gelu(A@W1 + b1); MODE2 out: ypair f32 = A@W2
// Block: 256 thr, tile 128(M)x64(N), BK=32. 8 waves in 4x2; each wave 32x32 -> 4 WMMA/step.
// Staging uses CDNA5 async global->LDS DMA, double-buffered.
template <int MODE>
__launch_bounds__(256)
__global__ void moe_ffn_wmma(const bf16* __restrict__ A, const bf16* __restrict__ Bt,
                             const float* __restrict__ bias, void* __restrict__ Out,
                             const int* __restrict__ offsets, const int* __restrict__ pair_token,
                             int Kdim, int N) {
    const int e   = blockIdx.z;
    const int off = offsets[e];
    const int n_e = offsets[e + 1] - off;
    const int mbase = blockIdx.y * 128;
    if (mbase >= n_e) return;
    const int nbase = blockIdx.x * 64;

    __shared__ __align__(16) bf16 As[2][128][40];   // M x K (+8 pad), double buffered
    __shared__ __align__(16) bf16 Bst[2][64][40];   // N x K (+8 pad), double buffered

    const int tid  = threadIdx.x;
    const int lane = tid & 31;
    const int w    = tid >> 5;
    const int half = lane >> 4;
    const int l15  = lane & 15;
    const int wm   = (w & 3) * 32;   // wave M strip (32 rows)
    const int wn   = (w >> 2) * 32;  // wave N strip (32 cols)

    const bf16* Bt_e = Bt + (size_t)e * (size_t)Kdim * (size_t)N;

    // staging: A tile 128x32 -> 2 thr/row x 16 elems; B tile 64x32 -> 4 thr/row x 8 elems
    const int ar = tid >> 1;            // 0..127
    const int ac = (tid & 1) * 16;      // 0,16
    int li = mbase + ar; if (li >= n_e) li = n_e - 1;
    int srcRow;
    if (MODE == 1) srcRow = pair_token[off + li];
    else           srcRow = off + li;
    const bf16* aptr = A + (size_t)srcRow * (size_t)Kdim + ac;

    const int br = tid >> 2;            // 0..63  (column n of B)
    const int bc = (tid & 3) * 8;       // 0..24
    const bf16* bptr = Bt_e + (size_t)(nbase + br) * (size_t)Kdim + bc;

    // per-thread LDS destinations for both buffers
    unsigned ldsA0[2], ldsA1[2], ldsB[2];
#pragma unroll
    for (int b = 0; b < 2; ++b) {
        ldsA0[b] = lds_off_of(&As[b][ar][ac]);
        ldsA1[b] = lds_off_of(&As[b][ar][ac + 8]);
        ldsB[b]  = lds_off_of(&Bst[b][br][bc]);
    }

    v8f c00 = {}, c01 = {}, c10 = {}, c11 = {};

    const int nk = Kdim / 32;
    // prologue: async-stage tile 0 directly into LDS
    async_ld_b128(ldsA0[0], aptr);
    async_ld_b128(ldsA1[0], aptr + 8);
    async_ld_b128(ldsB[0],  bptr);
    wait_async0();
    __syncthreads();

    for (int kt = 0; kt < nk; ++kt) {
        const int cur = kt & 1;
        const int nxt = 1 - cur;
        const int k1 = (kt + 1) * 32;
        const bool more = (kt + 1) < nk;
        if (more) {                       // DMA next tile while we compute this one
            async_ld_b128(ldsA0[nxt], aptr + k1);
            async_ld_b128(ldsA1[nxt], aptr + k1 + 8);
            async_ld_b128(ldsB[nxt],  bptr + k1);
            if (kt + 2 < nk) {
                __builtin_prefetch(aptr + k1 + 32, 0, 1);   // global_prefetch_b8
                __builtin_prefetch(bptr + k1 + 32, 0, 1);
            }
        }

        // fragments: per lane two ds_load_b128 each (ISA 7.12.2 layouts)
        FragU a0, a1, b0, b1;
        a0.q[0] = *(const uint4*)&As[cur][wm + l15][8 * half];        // K 0..7 (+8 upper half)
        a0.q[1] = *(const uint4*)&As[cur][wm + l15][16 + 8 * half];   // K 16..23 (+8)
        a1.q[0] = *(const uint4*)&As[cur][wm + 16 + l15][8 * half];
        a1.q[1] = *(const uint4*)&As[cur][wm + 16 + l15][16 + 8 * half];
        b0.q[0] = *(const uint4*)&Bst[cur][wn + l15][8 * half];       // K 0..7 (+8 upper half)
        b0.q[1] = *(const uint4*)&Bst[cur][wn + l15][16 + 8 * half];  // K 16..23 (+8)
        b1.q[0] = *(const uint4*)&Bst[cur][wn + 16 + l15][8 * half];
        b1.q[1] = *(const uint4*)&Bst[cur][wn + 16 + l15][16 + 8 * half];

        c00 = __builtin_amdgcn_wmma_f32_16x16x32_bf16(false, a0.v, false, b0.v, (short)0, c00, false, false);
        c01 = __builtin_amdgcn_wmma_f32_16x16x32_bf16(false, a0.v, false, b1.v, (short)0, c01, false, false);
        c10 = __builtin_amdgcn_wmma_f32_16x16x32_bf16(false, a1.v, false, b0.v, (short)0, c10, false, false);
        c11 = __builtin_amdgcn_wmma_f32_16x16x32_bf16(false, a1.v, false, b1.v, (short)0, c11, false, false);

        if (more) wait_async0();          // DMA for buffer nxt complete
        __syncthreads();                  // all waves done reading cur / writing nxt
    }

    // epilogue. C layout: VGPR v -> M = v (+8 lanes 16-31), N = lane&15
#pragma unroll
    for (int v = 0; v < 8; ++v) {
        const int r0 = mbase + wm + v + 8 * half;
        const int r1 = r0 + 16;
        const int col0 = nbase + wn + l15;
        const int col1 = col0 + 16;
        if (MODE == 1) {
            const float* be = bias + (size_t)e * (size_t)N;
            bf16* hout = (bf16*)Out;
            float bcol0 = be[col0], bcol1 = be[col1];
            if (r0 < n_e) {
                size_t orow = (size_t)(off + r0) * (size_t)N;
                float v0 = c00[v] + bcol0, v1 = c01[v] + bcol1;
                v0 = 0.5f * v0 * (1.0f + erff(v0 * 0.7071067811865475f));
                v1 = 0.5f * v1 * (1.0f + erff(v1 * 0.7071067811865475f));
                hout[orow + col0] = (bf16)v0;
                hout[orow + col1] = (bf16)v1;
            }
            if (r1 < n_e) {
                size_t orow = (size_t)(off + r1) * (size_t)N;
                float v0 = c10[v] + bcol0, v1 = c11[v] + bcol1;
                v0 = 0.5f * v0 * (1.0f + erff(v0 * 0.7071067811865475f));
                v1 = 0.5f * v1 * (1.0f + erff(v1 * 0.7071067811865475f));
                hout[orow + col0] = (bf16)v0;
                hout[orow + col1] = (bf16)v1;
            }
        } else {
            float* yout = (float*)Out;
            if (r0 < n_e) {
                size_t orow = (size_t)(off + r0) * (size_t)N;
                yout[orow + col0] = c00[v];
                yout[orow + col1] = c01[v];
            }
            if (r1 < n_e) {
                size_t orow = (size_t)(off + r1) * (size_t)N;
                yout[orow + col0] = c10[v];
                yout[orow + col1] = c11[v];
            }
        }
    }
}

// ---------------- combine: out[t] = w0*(y0 + b2[e0]) + w1*(y1 + b2[e1]) ----------------
__global__ void combine_kernel(const float* __restrict__ ypair, const float* __restrict__ b2,
                               const int* __restrict__ tok_e, const float* __restrict__ tok_w,
                               const int* __restrict__ pair_slot, float* __restrict__ out) {
    int t = blockIdx.x;
    int e0 = tok_e[2 * t], e1 = tok_e[2 * t + 1];
    float w0 = tok_w[2 * t], w1 = tok_w[2 * t + 1];
    int r0 = pair_slot[2 * t], r1 = pair_slot[2 * t + 1];
    const float* y0 = ypair + (size_t)r0 * DIM;
    const float* y1 = ypair + (size_t)r1 * DIM;
    const float* bb0 = b2 + (size_t)e0 * DIM;
    const float* bb1 = b2 + (size_t)e1 * DIM;
    float* o = out + (size_t)t * DIM;
    for (int d = threadIdx.x; d < DIM; d += blockDim.x)
        o[d] = w0 * (y0[d] + bb0[d]) + w1 * (y1[d] + bb1[d]);
}

// ---------------- launch ----------------
extern "C" void kernel_launch(void* const* d_in, const int* in_sizes, int n_in,
                              void* d_out, int out_size, void* d_ws, size_t ws_size,
                              hipStream_t stream) {
    (void)in_sizes; (void)n_in; (void)out_size; (void)ws_size;
    const float* x  = (const float*)d_in[0];
    const float* gw = (const float*)d_in[1];
    const float* w1 = (const float*)d_in[2];
    const float* b1 = (const float*)d_in[3];
    const float* w2 = (const float*)d_in[4];
    const float* b2 = (const float*)d_in[5];
    float* out = (float*)d_out;

    char* p = (char*)d_ws;
    auto alloc = [&](size_t bytes) -> char* {
        char* r = p; p += (bytes + 255) & ~(size_t)255; return r;
    };
    bf16*  xb         = (bf16*) alloc((size_t)BT_TOK * DIM * 2);
    bf16*  w1t        = (bf16*) alloc((size_t)NEXP * DIM * FDIM * 2);  // [E][F][D]
    bf16*  w2t        = (bf16*) alloc((size_t)NEXP * FDIM * DIM * 2);  // [E][D][F]
    bf16*  h          = (bf16*) alloc((size_t)NPAIR * FDIM * 2);
    float* ypair      = (float*)alloc((size_t)NPAIR * DIM * 4);
    int*   tok_e      = (int*)  alloc((size_t)NPAIR * 4);
    float* tok_w      = (float*)alloc((size_t)NPAIR * 4);
    int*   pair_token = (int*)  alloc((size_t)NPAIR * 4);
    int*   pair_slot  = (int*)  alloc((size_t)NPAIR * 4);
    int*   counts     = (int*)  alloc(64);
    int*   offsets    = (int*)  alloc(64);
    int*   cursor     = (int*)  alloc(64);

    init_counts_kernel<<<1, 32, 0, stream>>>(counts, cursor);

    { int n4 = BT_TOK * DIM / 4;
      cvt_f32_to_bf16<<<(n4 + 255) / 256, 256, 0, stream>>>(x, xb, n4); }

    // w1 [E][D][F] -> w1t [E][F][D]  (K=D rows, N=F cols)
    { dim3 g(FDIM / 32, DIM / 32, NEXP);
      transpose_cvt_kernel<<<g, 256, 0, stream>>>(w1, w1t, DIM, FDIM); }
    // w2 [E][F][D] -> w2t [E][D][F]  (K=F rows, N=D cols)
    { dim3 g(DIM / 32, FDIM / 32, NEXP);
      transpose_cvt_kernel<<<g, 256, 0, stream>>>(w2, w2t, FDIM, DIM); }

    gate_topk_kernel<<<BT_TOK / 8, 256, 0, stream>>>(x, gw, tok_e, tok_w, counts);
    prefix_kernel<<<1, 32, 0, stream>>>(counts, offsets);
    build_pairs_kernel<<<(BT_TOK + 255) / 256, 256, 0, stream>>>(tok_e, tok_w, offsets, cursor,
                                                                 pair_token, pair_slot);

    dim3 g1(FDIM / 64, BT_TOK / 128, NEXP);
    moe_ffn_wmma<1><<<g1, 256, 0, stream>>>(xb, w1t, b1, (void*)h, offsets, pair_token, DIM, FDIM);
    dim3 g2(DIM / 64, BT_TOK / 128, NEXP);
    moe_ffn_wmma<2><<<g2, 256, 0, stream>>>(h, w2t, nullptr, (void*)ypair, offsets, nullptr, FDIM, DIM);

    combine_kernel<<<BT_TOK, 256, 0, stream>>>(ypair, b2, tok_e, tok_w, pair_slot, out);
}
